// ResUpBlock_6073083756677
// MI455X (gfx1250) — matrix operand order repaired and checked by reading, not compiled, not run on hardware
//
#include <hip/hip_runtime.h>
#include <math.h>

typedef __attribute__((ext_vector_type(16))) __bf16 v16bf;
typedef __attribute__((ext_vector_type(8)))  float  v8f;

#define SQRT2F     1.41421356237309515f
#define INV_SQRT2F 0.70710678118654752f

__device__ __forceinline__ unsigned short f2bf(float f) {
  unsigned u = __float_as_uint(f);
  u += 0x7FFFu + ((u >> 16) & 1u);          // round-to-nearest-even
  return (unsigned short)(u >> 16);
}
__device__ __forceinline__ float ldval(const float* p, int i) { return p[i]; }
__device__ __forceinline__ float ldval(const unsigned short* p, int i) {
  return __uint_as_float(((unsigned)p[i]) << 16);
}

union FragB16 { uint4 u[2]; v16bf v; };

// ---- pack [Cout][Cin][KS][KS] f32 -> bf16 [Cout][KS*KS*Cin], k = (r*KS+s)*Cin + c ----
template <int Cin, int KS>
__global__ void pack_weights_kernel(const float* __restrict__ w, unsigned short* __restrict__ wp,
                                    float scale, int total) {
  int idx = blockIdx.x * blockDim.x + threadIdx.x;
  if (idx >= total) return;
  constexpr int kk = Cin * KS * KS;
  int m   = idx / kk;
  int rem = idx - m * kk;
  int c   = rem / (KS * KS);
  int rs  = rem - c * (KS * KS);
  wp[m * kk + rs * Cin + c] = f2bf(w[idx] * scale);
}

__device__ __forceinline__ float4 f4blend(float4 a, float4 b, float4 c, float4 d,
                                          float wa, float wb, float wc, float wd) {
  return make_float4(wa * a.x + wb * b.x + wc * c.x + wd * d.x,
                     wa * a.y + wb * b.y + wc * c.y + wd * d.y,
                     wa * a.z + wb * b.z + wc * c.z + wd * d.z,
                     wa * a.w + wb * b.w + wc * c.w + wd * d.w);
}

// ---- bilinear 2x upsample NHWC f32 -> NHWC bf16, 8 channels per thread, b128 in/out ----
template <int H, int W, int C>
__global__ void upsample2x_nhwc_kernel(const float* __restrict__ in,
                                       unsigned short* __restrict__ out, int total) {
  int idx = blockIdx.x * blockDim.x + threadIdx.x;
  if (idx >= total) return;                 // total = N*H2*W2*(C/8)
  constexpr int CG = C / 8, W2 = 2 * W, H2 = 2 * H;
  int cg = idx % CG;
  int w2 = (idx / CG) % W2;
  int h2 = (idx / (CG * W2)) % H2;
  int nI = idx / (CG * W2 * H2);
  float sh = h2 * 0.5f - 0.25f;
  float sw = w2 * 0.5f - 0.25f;
  int h0 = (int)floorf(sh); float fh = sh - h0;
  int w0 = (int)floorf(sw); float fw = sw - w0;
  int h0c = max(h0, 0), h1c = min(h0 + 1, H - 1);
  int w0c = max(w0, 0), w1c = min(w0 + 1, W - 1);
  float w00 = (1.f - fh) * (1.f - fw), w01 = (1.f - fh) * fw;
  float w10 = fh * (1.f - fw),         w11 = fh * fw;
  const float* base = in + (size_t)nI * H * W * C + cg * 8;
  const float* p00 = base + ((size_t)h0c * W + w0c) * C;
  const float* p01 = base + ((size_t)h0c * W + w1c) * C;
  const float* p10 = base + ((size_t)h1c * W + w0c) * C;
  const float* p11 = base + ((size_t)h1c * W + w1c) * C;
  float4 lo = f4blend(*(const float4*)p00, *(const float4*)p01,
                      *(const float4*)p10, *(const float4*)p11, w00, w01, w10, w11);
  float4 hi = f4blend(*(const float4*)(p00 + 4), *(const float4*)(p01 + 4),
                      *(const float4*)(p10 + 4), *(const float4*)(p11 + 4), w00, w01, w10, w11);
  uint4 r;
  r.x = (unsigned)f2bf(lo.x) | ((unsigned)f2bf(lo.y) << 16);
  r.y = (unsigned)f2bf(lo.z) | ((unsigned)f2bf(lo.w) << 16);
  r.z = (unsigned)f2bf(hi.x) | ((unsigned)f2bf(hi.y) << 16);
  r.w = (unsigned)f2bf(hi.z) | ((unsigned)f2bf(hi.w) << 16);
  *(uint4*)(out + (size_t)idx * 8) = r;
}

// ---- implicit-GEMM conv via WMMA bf16, double-buffered LDS pipeline ----
// Block: 256 thr (8 waves) -> 128(M) x 64(Npix); wave = 16(M) x 64(Npix), 4 accumulators.
// IN_NHWC:  input bf16 channel-contiguous -> 1 b128 load/thread/K-step.
// OUT_NHWC: epilogue writes 2x float4 per lane (8 consecutive channels).
// EPI: 0 = plain store; 1 = +bias, lrelu*sqrt2; 2 = EPI1 then + bilinear-up(skip NCHW) and /sqrt2
template <typename TIN, int KS, int EPI, bool IN_NHWC, bool OUT_NHWC,
          int Cin, int Cout, int H, int W, int HS, int WS>
__global__ __launch_bounds__(256) void conv_wmma_kernel(
    const TIN* __restrict__ in, const unsigned short* __restrict__ wpack,
    const float* __restrict__ bias, const float* __restrict__ skip,
    float* __restrict__ out) {
  constexpr int Ktot   = Cin * KS * KS;
  constexpr int KTILES = Ktot / 32;
  constexpr int CC     = Cin / 32;
  constexpr int HW     = H * W;
  __shared__ __align__(16) unsigned short sA[2][128][40];  // [buf][M][K] +8 pad
  __shared__ __align__(16) unsigned short sB[2][32][72];   // [buf][K][px] +8 pad

  const int tid  = threadIdx.x;
  const int lane = tid & 31;
  const int wave = tid >> 5;
  const int n0p  = blockIdx.x * 64;   // pixel tile base; 64 | W -> tile within one image row
  const int m0b  = blockIdx.y * 128;  // cout tile base

  const int n     = n0p / HW;         // compile-time shifts
  const int h     = (n0p / W) % H;
  const int wbase = n0p % W;
  const int px2   = lane << 1;        // NCHW loader: pixel pair
  const int pxl   = tid & 63;         // NHWC loader: single pixel
  const int cg    = tid >> 6;         // NHWC loader: channel group (8 ch)

  const int arowl = tid >> 1;         // weight-stage row owned by this thread
  const int acolb = (tid & 1) << 4;

  struct StageRegs { uint4 wa0, wa1; uint4 nb; unsigned bp[4]; };

  auto stageG = [&](int kt, StageRegs& sr) {
    const int rs  = (KS == 1) ? 0 : (kt / CC);
    const int cc  = (KS == 1) ? kt : (kt % CC);
    const int dr  = (KS == 1) ? 0 : (rs / 3 - 1);
    const int dsS = (KS == 1) ? 0 : (rs % 3 - 1);
    const int hh  = h + dr;
    const bool hok = (KS == 1) || ((unsigned)hh < (unsigned)H);
    const int  k0 = rs * Cin + cc * 32;
    const uint4* src = (const uint4*)(wpack + (size_t)(m0b + arowl) * Ktot + k0 + acolb);
    sr.wa0 = src[0];
    sr.wa1 = src[1];
    if constexpr (IN_NHWC) {
      const int wwp = wbase + pxl + dsS;
      const bool okp = hok && ((KS == 1) || ((unsigned)wwp < (unsigned)W));
      sr.nb = make_uint4(0u, 0u, 0u, 0u);
      if (okp)
        sr.nb = *(const uint4*)(in + (size_t)((n * H + hh) * W + wwp) * Cin + cc * 32 + (cg << 3));
    } else {
      const int  ww0 = wbase + px2 + dsS;
      const bool ok0 = hok && ((KS == 1) || ((unsigned)ww0 < (unsigned)W));
      const bool ok1 = hok && ((KS == 1) || ((unsigned)(ww0 + 1) < (unsigned)W));
      const int  gbase = n * Cin * HW + hh * W + ww0;   // guarded by ok0/ok1 when hh==-1
#pragma unroll
      for (int i = 0; i < 4; ++i) {
        int kk = wave + (i << 3);
        int gi = gbase + (cc * 32 + kk) * HW;
        unsigned p0 = (unsigned)f2bf(ok0 ? ldval(in, gi)     : 0.f);
        unsigned p1 = (unsigned)f2bf(ok1 ? ldval(in, gi + 1) : 0.f);
        sr.bp[i] = p0 | (p1 << 16);
      }
    }
  };
  auto stageL = [&](int buf, const StageRegs& sr) {
    *(uint4*)&sA[buf][arowl][acolb]     = sr.wa0;
    *(uint4*)&sA[buf][arowl][acolb + 8] = sr.wa1;
    if constexpr (IN_NHWC) {
      const unsigned short* e = (const unsigned short*)&sr.nb;
#pragma unroll
      for (int j = 0; j < 8; ++j) sB[buf][(cg << 3) + j][pxl] = e[j];
    } else {
#pragma unroll
      for (int i = 0; i < 4; ++i) *(unsigned*)&sB[buf][wave + (i << 3)][px2] = sr.bp[i];
    }
  };

  v8f acc[4] = {};
  StageRegs sr{};
  stageG(0, sr);
  stageL(0, sr);
  __syncthreads();

  for (int kt = 0; kt < KTILES; ++kt) {
    const int cur = kt & 1;
    StageRegs nx{};
    if (kt + 1 < KTILES) stageG(kt + 1, nx);              // global loads overlap WMMAs
    {
      FragB16 a;
      int arow = (wave << 4) + (lane & 15);
      int aoff = (lane & 16) >> 1;                        // lanes>=16 take K+8 halves
      a.u[0] = *(const uint4*)&sA[cur][arow][aoff];
      a.u[1] = *(const uint4*)&sA[cur][arow][16 + aoff];
#pragma unroll
      for (int t = 0; t < 4; ++t) {
        FragB16 b;                                        // lane = N col, elements = K (table order)
        unsigned short* be = (unsigned short*)&b;
#pragma unroll
        for (int s = 0; s < 16; ++s) be[s] = sB[cur][(s < 8 ? s : s + 8) + aoff][(t << 4) + (lane & 15)];
        acc[t] = __builtin_amdgcn_wmma_f32_16x16x32_bf16(false, a.v, false, b.v,
                                                         (short)0, acc[t], false, false);
      }
    }
    if (kt + 1 < KTILES) stageL(cur ^ 1, nx);             // fill other buffer
    __syncthreads();                                      // single barrier per K-step
  }

  // --- epilogue: C/D layout lane=N(pixel), VGPR r = M=r (+8 for lanes>=16) ---
  const int m0wv = m0b + (wave << 4);
  float fh = 0.f; int h0c = 0, h1c = 0;
  if (EPI == 2) {                                         // h is block-uniform
    float sh = h * 0.5f - 0.25f;
    int h0 = (int)floorf(sh); fh = sh - h0;
    h0c = max(h0, 0); h1c = min(h0 + 1, HS - 1);
  }
#pragma unroll
  for (int t = 0; t < 4; ++t) {
    if constexpr (OUT_NHWC) {
      int p = n0p + (t << 4) + (lane & 15);               // flat pixel, NHWC
      float v[8];
#pragma unroll
      for (int r = 0; r < 8; ++r) {
        int cout = m0wv + r + ((lane & 16) >> 1);
        float x = acc[t][r];
        if (EPI >= 1) {
          x += bias[cout];
          x = (x >= 0.f ? x : 0.2f * x) * SQRT2F;
        }
        v[r] = x;
      }
      float* dst = out + (size_t)p * Cout + m0wv + ((lane & 16) >> 1);
      *(float4*)dst       = make_float4(v[0], v[1], v[2], v[3]);
      *(float4*)(dst + 4) = make_float4(v[4], v[5], v[6], v[7]);
    } else {
      int w = wbase + (t << 4) + (lane & 15);
      float fw = 0.f; int w0c = 0, w1c = 0;
      if (EPI == 2) {
        float sw = w * 0.5f - 0.25f;
        int w0 = (int)floorf(sw); fw = sw - w0;
        w0c = max(w0, 0); w1c = min(w0 + 1, WS - 1);
      }
#pragma unroll
      for (int r = 0; r < 8; ++r) {
        int cout = m0wv + r + ((lane & 16) >> 1);
        float v = acc[t][r];
        if (EPI >= 1) {
          v += bias[cout];
          v = (v >= 0.f ? v : 0.2f * v) * SQRT2F;
        }
        if (EPI == 2) {
          const float* sb = skip + (size_t)(n * Cout + cout) * HS * WS;
          float sv = (1.f - fh) * ((1.f - fw) * sb[h0c * WS + w0c] + fw * sb[h0c * WS + w1c]) +
                            fh  * ((1.f - fw) * sb[h1c * WS + w0c] + fw * sb[h1c * WS + w1c]);
          v = (v + sv) * INV_SQRT2F;
        }
        out[((size_t)(n * Cout + cout) * H + h) * W + w] = v;
      }
    }
  }
}

extern "C" void kernel_launch(void* const* d_in, const int* in_sizes, int n_in,
                              void* d_out, int out_size, void* d_ws, size_t ws_size,
                              hipStream_t stream) {
  const float* x   = (const float*)d_in[0];   // [8,256,64,64] NCHW
  const float* w1  = (const float*)d_in[1];   // [256,256,3,3]
  const float* b1  = (const float*)d_in[2];   // [256]
  const float* w2  = (const float*)d_in[3];   // [128,256,3,3]
  const float* b2  = (const float*)d_in[4];   // [128]
  const float* wsk = (const float*)d_in[5];   // [128,256,1,1]
  float* out = (float*)d_out;                 // [8,128,128,128] NCHW

  const int N = 8;

  char* ws = (char*)d_ws;
  float*          A   = (float*)(ws);                          // conv1 out f32 NHWC [8,64,64,256]   33.55 MB
  float*          S   = (float*)(ws + 33554432);               // skip 1x1 f32 NCHW [8,128,64,64]    16.78 MB
  unsigned short* B   = (unsigned short*)(ws + 50331648);      // up(A) bf16 NHWC [8,128,128,256]    67.11 MB
  unsigned short* W1p = (unsigned short*)(ws + 117440512);     // 256x2304 bf16
  unsigned short* W2p = (unsigned short*)(ws + 118620160);     // 128x2304 bf16
  unsigned short* WSp = (unsigned short*)(ws + 119209984);     // 128x256  bf16

  // pack + scale weights (equalized-lr scales: 1/sqrt(256*9)=1/48, 1/sqrt(256)=1/16)
  { int t = 256 * 256 * 9;
    pack_weights_kernel<256, 3><<<(t + 255) / 256, 256, 0, stream>>>(w1, W1p, 1.0f / 48.0f, t); }
  { int t = 128 * 256 * 9;
    pack_weights_kernel<256, 3><<<(t + 255) / 256, 256, 0, stream>>>(w2, W2p, 1.0f / 48.0f, t); }
  { int t = 128 * 256;
    pack_weights_kernel<256, 1><<<(t + 255) / 256, 256, 0, stream>>>(wsk, WSp, 1.0f / 16.0f, t); }

  // conv1: 3x3 256->256 @64x64 (NCHW in), bias+lrelu*sqrt2 -> A (NHWC)
  conv_wmma_kernel<float, 3, 1, false, true, 256, 256, 64, 64, 1, 1>
      <<<dim3(N * 64 * 64 / 64, 256 / 128), 256, 0, stream>>>(x, W1p, b1, nullptr, A);

  // skip (commuted): 1x1 256->128 @64x64 -> S (NCHW; upsampled in conv2 epilogue)
  conv_wmma_kernel<float, 1, 0, false, false, 256, 128, 64, 64, 1, 1>
      <<<dim3(N * 64 * 64 / 64, 128 / 128), 256, 0, stream>>>(x, WSp, nullptr, nullptr, S);

  // bilinear up 2x: A (NHWC f32) -> B (NHWC bf16) @128x128
  { int tot = N * 128 * 128 * (256 / 8);
    upsample2x_nhwc_kernel<64, 64, 256><<<tot / 256, 256, 0, stream>>>(A, B, tot); }

  // conv2: 3x3 256->128 @128x128 (NHWC bf16 in), bias+lrelu*sqrt2, + up(S), /sqrt2 -> out (NCHW)
  conv_wmma_kernel<unsigned short, 3, 2, true, false, 256, 128, 128, 128, 64, 64>
      <<<dim3(N * 128 * 128 / 64, 128 / 128), 256, 0, stream>>>(B, W2p, b2, S, out);
}
